// CrossAttention_78271484002687
// MI455X (gfx1250) — matrix operand order repaired and checked by reading, not compiled, not run on hardware
//
#include <hip/hip_runtime.h>

// MI455X / gfx1250, wave32. Fused hard-attention cross-attention:
//   k = x@w_ks^T, v = x@w_vs^T, attn = q@k^T (scale folded, argmax-invariant),
//   idx = argmax(attn), hardattn = one_hot(idx), acc[b][idx[n]] += v[n],
//   out = acc @ w_fc^T.
// HBM-bound: ~134 MB x read + ~134 MB one-hot write => ~11.6 us floor @ 23.3 TB/s.
// All big GEMMs use v_wmma_f32_16x16x32_bf16; attn scores never touch HBM.

typedef __bf16 bf16x16 __attribute__((ext_vector_type(16)));
typedef float  floatx8 __attribute__((ext_vector_type(8)));
typedef float  floatx4 __attribute__((ext_vector_type(4)));

#define WMMA_BF16(A, B, C) \
  __builtin_amdgcn_wmma_f32_16x16x32_bf16(false, (A), false, (B), (short)0, (C), false, false)

constexpr int kB  = 16;
constexpr int kN  = 8192;
constexpr int kDM = 256;
constexpr int kDK = 64;
constexpr int kDV = 64;
constexpr int kNQ = 256;

// LDS: weight/q fragments (shared, pre-permuted to per-lane B-operand order) +
// per-wave k-projection spill. ~112.5 KB -> 2 workgroups per WGP (320 KB LDS).
constexpr int kLdsWFrag = 8 * 4 * 32 * 16;    // [kc][t][lane][e] bf16
constexpr int kLdsQFrag = 2 * 16 * 32 * 16;   // [kc][t][lane][e] bf16
constexpr int kLdsK     = 8 * 16 * 64;        // per-wave 16x64 bf16 k-proj spill
constexpr size_t kLdsBytes =
    (size_t)(2 * kLdsWFrag + kLdsQFrag + kLdsK) * sizeof(__bf16);

__global__ __launch_bounds__(256, 1) void fused_hardattn_kernel(
    const float* __restrict__ x, const float* __restrict__ qm,
    const float* __restrict__ wks, const float* __restrict__ wvs,
    float* __restrict__ hard, float* __restrict__ acc)
{
  extern __shared__ char smem_raw[];
  __bf16* wkF = reinterpret_cast<__bf16*>(smem_raw);
  __bf16* wvF = wkF + kLdsWFrag;
  __bf16* qF  = wvF + kLdsWFrag;
  __bf16* kst = qF + kLdsQFrag;

  const int tid  = threadIdx.x;
  const int lane = tid & 31;
  const int wv   = tid >> 5;
  const int lo   = lane & 15;
  const int hi8  = (lane >> 4) << 3;   // 0 or 8
  const int b    = blockIdx.y;
  const int n0   = (blockIdx.x * 8 + wv) * 16;   // this wave's 16 token rows

  // ---- Phase 1: weights -> bf16 B-matrix fragments in LDS (frag order:
  //      lane holds column N=lane&15, K = (e&7) + 8*(lane>=16) + 16*(e>=8); 32B contiguous/lane)
  for (int i = tid; i < kLdsWFrag; i += 256) {
    const int e = i & 15, ln = (i >> 4) & 31, t = (i >> 9) & 3, kc = i >> 11;
    const int K = kc * 32 + (e & 7) + ((ln >> 4) << 3) + ((e >> 3) << 4);
    const int n = t * 16 + (ln & 15);
    wkF[i] = (__bf16)wks[n * kDM + K];
    wvF[i] = (__bf16)wvs[n * kDM + K];
  }
  for (int i = tid; i < kLdsQFrag; i += 256) {
    const int e = i & 15, ln = (i >> 4) & 31, t = (i >> 9) & 15, kc = i >> 13;
    const int K = kc * 32 + (e & 7) + ((ln >> 4) << 3) + ((e >> 3) << 4);
    const int n = t * 16 + (ln & 15);
    // fold scale 1/sqrt(NQ)=1/16 into q (argmax is invariant to it anyway)
    qF[i] = (__bf16)(qm[n * kDK + K] * 0.0625f);
  }
  __syncthreads();

  // ---- Phase 2: K and V projections (16x64 each), contraction over d_model=256.
  //      A-fragments loaded straight from global x (each 128B line is exactly one
  //      32-K chunk of one row -> every fetched byte consumed), f32->bf16 in VALU.
  const float* arow = x + ((size_t)(b * kN) + n0 + lo) * kDM + hi8;
  floatx8 kacc[4] = {}, vacc[4] = {};
#pragma unroll
  for (int kc = 0; kc < 8; ++kc) {
    // 16-bit A 16x32 layout: lanes<16 hold K in [0..7]u[16..23]; lanes>=16: +8
    const floatx4* p = reinterpret_cast<const floatx4*>(arow + kc * 32);
    const floatx4 f0 = p[0], f1 = p[1], f2 = p[4], f3 = p[5];
    bf16x16 a;
#pragma unroll
    for (int e = 0; e < 4; ++e) {
      a[e]      = (__bf16)f0[e];
      a[e + 4]  = (__bf16)f1[e];
      a[e + 8]  = (__bf16)f2[e];
      a[e + 12] = (__bf16)f3[e];
    }
#pragma unroll
    for (int t = 0; t < 4; ++t) {
      const bf16x16 bk = *reinterpret_cast<const bf16x16*>(wkF + ((kc * 4 + t) * 32 + lane) * 16);
      kacc[t] = WMMA_BF16(a, bk, kacc[t]);
      const bf16x16 bv = *reinterpret_cast<const bf16x16*>(wvF + ((kc * 4 + t) * 32 + lane) * 16);
      vacc[t] = WMMA_BF16(a, bv, vacc[t]);
    }
  }

  // ---- Phase 3: spill k-projection to LDS (bf16) so it can be reloaded as A
  __bf16* myk = kst + wv * (16 * 64);
#pragma unroll
  for (int t = 0; t < 4; ++t)
#pragma unroll
    for (int r = 0; r < 8; ++r)
      myk[(r + hi8) * 64 + t * 16 + lo] = (__bf16)kacc[t][r];   // C/D: M=r+hi8, N=lo
  __syncthreads();

  // ---- Phase 4: attn = kproj @ q^T (16x256), streaming argmax (attn stays in VGPRs)
  bf16x16 a0, a1;
  {
    union { uint4 u[2]; bf16x16 v; } af;
    const uint4* ap0 = reinterpret_cast<const uint4*>(myk + lo * 64 + hi8);
    af.u[0] = ap0[0]; af.u[1] = ap0[2]; a0 = af.v;
    const uint4* ap1 = reinterpret_cast<const uint4*>(myk + lo * 64 + 32 + hi8);
    af.u[0] = ap1[0]; af.u[1] = ap1[2]; a1 = af.v;
  }
  float bestv[8]; int besti[8];
#pragma unroll
  for (int r = 0; r < 8; ++r) { bestv[r] = -__builtin_inff(); besti[r] = 0; }
  for (int t = 0; t < 16; ++t) {
    floatx8 c = {};
    c = WMMA_BF16(a0, *reinterpret_cast<const bf16x16*>(qF + (t * 32 + lane) * 16), c);
    c = WMMA_BF16(a1, *reinterpret_cast<const bf16x16*>(qF + ((16 + t) * 32 + lane) * 16), c);
    const int nq = t * 16 + lo;
#pragma unroll
    for (int r = 0; r < 8; ++r)
      if (c[r] > bestv[r]) { bestv[r] = c[r]; besti[r] = nq; }   // strict >: keep first (lowest nq)
  }
  // reduce (max, lowest-index-on-tie) across the 16 lanes holding each row
#pragma unroll
  for (int r = 0; r < 8; ++r) {
    float v = bestv[r]; int ix = besti[r];
#pragma unroll
    for (int d = 1; d < 16; d <<= 1) {
      const float ov = __shfl_xor(v, d, 32);
      const int   oi = __shfl_xor(ix, d, 32);
      if (ov > v || (ov == v && oi < ix)) { v = ov; ix = oi; }
    }
    besti[r] = ix;   // final winner for row r+hi8, uniform across the 16-lane half
  }
  // broadcast all 16 row winners into registers (lane0 holds rows 0..7, lane16 rows 8..15)
  int rowidx[16];
#pragma unroll
  for (int r = 0; r < 8; ++r) {
    rowidx[r]     = __shfl(besti[r], 0, 32);
    rowidx[r + 8] = __shfl(besti[r], 16, 32);
  }

  // ---- Phase 5: one-hot hardattn rows, vectorized nontemporal b128 stream (134 MB)
  float* hrow = hard + ((size_t)b * kN + n0) * kNQ;
#pragma unroll
  for (int r = 0; r < 16; ++r) {
    const int target = rowidx[r];
#pragma unroll
    for (int j = 0; j < 2; ++j) {
      const int nq0 = j * 128 + lane * 4;
      floatx4 vv;
      vv[0] = (nq0 + 0 == target) ? 1.0f : 0.0f;
      vv[1] = (nq0 + 1 == target) ? 1.0f : 0.0f;
      vv[2] = (nq0 + 2 == target) ? 1.0f : 0.0f;
      vv[3] = (nq0 + 3 == target) ? 1.0f : 0.0f;
      __builtin_nontemporal_store(vv, reinterpret_cast<floatx4*>(hrow + r * kNQ + nq0));
    }
  }

  // ---- Phase 6: scatter v rows into acc[b][slot][dv]. Lane already holds the
  //      winning slot (besti[r]) for exactly the rows whose v-values it owns.
#pragma unroll
  for (int t = 0; t < 4; ++t)
#pragma unroll
    for (int r = 0; r < 8; ++r)
      atomicAdd(acc + ((size_t)b * kNQ + besti[r]) * kDV + t * 16 + lo, vacc[t][r]);
}

// out[b][qi][dm] = sum_dv acc[b][qi][dv] * wfc[dm][dv] ; one 16x16 tile per wave.
__global__ __launch_bounds__(256, 1) void fc_proj_kernel(
    const float* __restrict__ acc, const float* __restrict__ wfc,
    float* __restrict__ out)
{
  const int lane = threadIdx.x & 31, wv = threadIdx.x >> 5;
  const int lo = lane & 15, hi8 = (lane >> 4) << 3, hi16 = (lane >> 4) << 4;
  const int tile = blockIdx.x * 8 + wv;             // 16 b * 16 qt * 16 dt = 4096 tiles
  const int b = tile >> 8, qt = (tile >> 4) & 15, dt = tile & 15;
  const float* A = acc + ((size_t)b * kNQ + qt * 16) * kDV;
  floatx8 c = {};
#pragma unroll
  for (int kc = 0; kc < 2; ++kc) {
    bf16x16 a, bb;
    const int ro = lo * kDV + kc * 32 + hi16;
#pragma unroll
    for (int e = 0; e < 8; ++e) {
      a[e]     = (__bf16)A[ro + e];
      a[e + 8] = (__bf16)A[ro + 8 + e];
    }
    const int dm = dt * 16 + lo;
    const int kb = kc * 32 + hi16;
#pragma unroll
    for (int e = 0; e < 16; ++e)
      bb[e] = (__bf16)wfc[dm * kDV + kb + e];       // B[k=dv][n=dm] = wfc[dm][dv]
    c = WMMA_BF16(a, bb, c);
  }
  float* o = out + ((size_t)b * kNQ + qt * 16) * kDM + dt * 16 + lo;
#pragma unroll
  for (int r = 0; r < 8; ++r)
    o[(r + hi8) * kDM] = c[r];
}

extern "C" void kernel_launch(void* const* d_in, const int* in_sizes, int n_in,
                              void* d_out, int out_size, void* d_ws, size_t ws_size,
                              hipStream_t stream) {
  (void)in_sizes; (void)n_in; (void)out_size; (void)ws_size;
  const float* x   = (const float*)d_in[0];   // (16, 8192, 256)
  const float* q   = (const float*)d_in[1];   // (256, 64)
  const float* wks = (const float*)d_in[2];   // (64, 256)
  const float* wvs = (const float*)d_in[3];   // (64, 256)
  const float* wfc = (const float*)d_in[4];   // (256, 64)

  float* out  = (float*)d_out;                          // (16, 256, 256)
  float* hard = out + (size_t)kB * kNQ * kDM;           // (16, 8192, 256)
  float* acc  = (float*)d_ws;                           // (16, 256, 64) f32 accumulator

  hipMemsetAsync(acc, 0, (size_t)kB * kNQ * kDV * sizeof(float), stream);

  dim3 grid(kN / 128, kB);   // 8 waves/block * 16 rows/wave = 128 rows/block
  fused_hardattn_kernel<<<grid, 256, kLdsBytes, stream>>>(x, q, wks, wvs, hard, acc);

  fc_proj_kernel<<<(kB * 16 * 16) / 8, 256, 0, stream>>>(acc, wfc, out);
}
